// GRUClassifier_22282290332128
// MI455X (gfx1250) — compile-verified, matrix operand order checked by reading, not run on hardware
//
#include <hip/hip_runtime.h>
#include <math.h>

// Problem constants (match reference)
#define Bc   512
#define Tt   256
#define ISc  128
#define Hc   256
#define H2c  512
#define G3   768   // 3 * H gates

typedef __bf16 bf16_t;
typedef __attribute__((ext_vector_type(16))) __bf16 v16bf;
typedef __attribute__((ext_vector_type(8)))  float  v8f;

// ---------------- scalar helpers ----------------
__device__ inline float bf2f(bf16_t b) {
  union { unsigned short s; bf16_t b; } i; i.b = b;
  union { unsigned u; float f; } o; o.u = ((unsigned)i.s) << 16; return o.f;
}
__device__ inline bf16_t f2bf(float f) {
  union { float f; unsigned u; } v; v.f = f;
  unsigned r = (v.u + 0x7FFFu + ((v.u >> 16) & 1u)) >> 16;
  union { unsigned short s; bf16_t b; } o; o.s = (unsigned short)r; return o.b;
}
__device__ inline float sigmoidf_(float x) { return 1.f / (1.f + expf(-x)); }
__device__ inline float geluf_(float x)    { return 0.5f * x * (1.f + erff(x * 0.70710678118f)); }
__device__ inline v8f vzero() { v8f v = {0.f,0.f,0.f,0.f,0.f,0.f,0.f,0.f}; return v; }

// Load one 16x32 bf16 WMMA operand fragment (per-lane layout per CDNA5 ISA:
// lane&15 selects row/col; lane>>4 selects K-half; elements 0..7 -> K=kb..kb+7,
// elements 8..15 -> K=kb+16..kb+23). Two 16B loads per fragment.
__device__ inline v16bf frag16(const bf16_t* p) {
  union { v16bf v; uint4 q[2]; } u;
  u.q[0] = *reinterpret_cast<const uint4*>(p);
  u.q[1] = *reinterpret_cast<const uint4*>(p + 16);
  return u.v;
}

// Accumulate a 32x32 output tile: A = sA rows (LDS, [32+..][K]), B = W rows
// treated as B^T columns (global, row-major [n][K] -> contiguous K per lane).
__device__ inline void mma_block(v8f (&acc)[2][2], const bf16_t* sA,
                                 const bf16_t* __restrict__ W, int K, int lane) {
  const int lr = lane & 15;
  const int kb = (lane >> 4) << 3;
  for (int k0 = 0; k0 < K; k0 += 32) {
    if (k0 + 32 < K) {  // near-scope (WGP$) prefetch of next weight fragments
      __builtin_prefetch(W + (size_t)lr * K + (k0 + 32) + kb, 0, 3);
      __builtin_prefetch(W + (size_t)(16 + lr) * K + (k0 + 32) + kb, 0, 3);
    }
    v16bf a0 = frag16(sA + (size_t)lr * K + k0 + kb);
    v16bf a1 = frag16(sA + (size_t)(16 + lr) * K + k0 + kb);
    v16bf b0 = frag16(W + (size_t)lr * K + k0 + kb);
    v16bf b1 = frag16(W + (size_t)(16 + lr) * K + k0 + kb);
    acc[0][0] = __builtin_amdgcn_wmma_f32_16x16x32_bf16(false, a0, false, b0, (short)0, acc[0][0], false, false);
    acc[0][1] = __builtin_amdgcn_wmma_f32_16x16x32_bf16(false, a0, false, b1, (short)0, acc[0][1], false, false);
    acc[1][0] = __builtin_amdgcn_wmma_f32_16x16x32_bf16(false, a1, false, b0, (short)0, acc[1][0], false, false);
    acc[1][1] = __builtin_amdgcn_wmma_f32_16x16x32_bf16(false, a1, false, b1, (short)0, acc[1][1], false, false);
  }
}

// ---------------- kernels ----------------

__global__ void to_bf16_kernel(const float* __restrict__ s, bf16_t* __restrict__ d, int n) {
  int i = blockIdx.x * blockDim.x + threadIdx.x;
  if (i < n) d[i] = f2bf(s[i]);
}

// encoded = mish(x @ enc_w.T + enc_b); write seq as [T][B][IS] bf16
__global__ void encoder_kernel(const float* __restrict__ x, const float* __restrict__ enc_w,
                               const float* __restrict__ enc_b, bf16_t* __restrict__ seqbf) {
  size_t idx = (size_t)blockIdx.x * blockDim.x + threadIdx.x;
  if (idx >= (size_t)Bc * Tt * ISc) return;
  int i = (int)(idx % ISc);
  size_t r = idx / ISc;
  int t = (int)(r % Tt);
  int b = (int)(r / Tt);
  int n = t * ISc + i;
  const float* wrow = enc_w + (size_t)n * 64;
  const float* xr   = x + (size_t)b * 64;
  float acc = enc_b[n];
#pragma unroll
  for (int k = 0; k < 64; ++k) acc += xr[k] * wrow[k];
  float sp = (acc > 20.f) ? acc : log1pf(expf(acc));
  seqbf[((size_t)t * Bc + b) * ISc + i] = f2bf(acc * tanhf(sp));
}

// One (layer, timestep): both directions of the GRU cell, fully fused.
// grid = (8 btiles, 4 jtiles, 2 dirs), block = 128 (4 wave32, 2x2 over 64x64).
__global__ void gru_layer_kernel(const bf16_t* __restrict__ inp, int Kin,      // [B][Kin]
                                 const bf16_t* __restrict__ Wih,               // [2][768][Kin]
                                 const bf16_t* __restrict__ Whh,               // [2][768][256]
                                 const float*  __restrict__ bih,               // [2][768]
                                 const float*  __restrict__ bhh,               // [2][768]
                                 const bf16_t* __restrict__ hbf_in,            // [2][B][256]
                                 const float*  __restrict__ h32_in,            // [2][B][256]
                                 float*  __restrict__ h32_out,
                                 bf16_t* __restrict__ hbf_out,
                                 bf16_t* __restrict__ lh_out) {                // [B][512]
  const int dir  = blockIdx.z;
  const int b0   = blockIdx.x * 64;
  const int j0   = blockIdx.y * 64;
  const int tid  = threadIdx.x;
  const int lane = tid & 31;
  const int wave = tid >> 5;
  const int wr   = (wave & 1) * 32;   // wave row offset in 64x64 tile
  const int wc   = (wave >> 1) * 32;  // wave col offset

  extern __shared__ uint4 dynsm[];
  bf16_t* sInp = (bf16_t*)dynsm;          // [64][Kin]
  bf16_t* sH   = sInp + (size_t)64 * Kin; // [64][256]

  { // cooperative staging (contiguous row blocks -> flat 16B copies)
    const uint4* gsrc = reinterpret_cast<const uint4*>(inp + (size_t)b0 * Kin);
    uint4* ldst = reinterpret_cast<uint4*>(sInp);
    const int n16 = 64 * Kin / 8;
    for (int i = tid; i < n16; i += blockDim.x) ldst[i] = gsrc[i];
    const uint4* hsrc = reinterpret_cast<const uint4*>(hbf_in + ((size_t)dir * Bc + b0) * Hc);
    uint4* hdst = reinterpret_cast<uint4*>(sH);
    for (int i = tid; i < 64 * Hc / 8; i += blockDim.x) hdst[i] = hsrc[i];
  }
  __syncthreads();

  const bf16_t* sAw = sInp + (size_t)wr * Kin;
  const bf16_t* sHw = sH   + (size_t)wr * Hc;
  const bf16_t* Wd  = Wih + (size_t)dir * G3 * Kin;
  const bf16_t* Ud  = Whh + (size_t)dir * G3 * Hc;
  const float*  bi  = bih + (size_t)dir * G3;
  const float*  bh  = bhh + (size_t)dir * G3;
  const int nbase   = j0 + wc;            // gate-lane column base (0..255)
  const int jlane   = lane & 15;

  v8f rr[2][2], zz[2][2];

  { // r = sigmoid(ir + hr + b)
    v8f gi[2][2] = {vzero(),vzero(),vzero(),vzero()};
    v8f gh[2][2] = {vzero(),vzero(),vzero(),vzero()};
    mma_block(gi, sAw, Wd + (size_t)(0 * Hc + nbase) * Kin, Kin, lane);
    mma_block(gh, sHw, Ud + (size_t)(0 * Hc + nbase) * Hc,  Hc,  lane);
#pragma unroll
    for (int mi = 0; mi < 2; ++mi)
#pragma unroll
      for (int ni = 0; ni < 2; ++ni) {
        int j = j0 + wc + ni * 16 + jlane;
        float bsum = bi[j] + bh[j];
#pragma unroll
        for (int e = 0; e < 8; ++e)
          rr[mi][ni][e] = sigmoidf_(gi[mi][ni][e] + gh[mi][ni][e] + bsum);
      }
  }
  { // z = sigmoid(iz + hz + b)
    v8f gi[2][2] = {vzero(),vzero(),vzero(),vzero()};
    v8f gh[2][2] = {vzero(),vzero(),vzero(),vzero()};
    mma_block(gi, sAw, Wd + (size_t)(1 * Hc + nbase) * Kin, Kin, lane);
    mma_block(gh, sHw, Ud + (size_t)(1 * Hc + nbase) * Hc,  Hc,  lane);
#pragma unroll
    for (int mi = 0; mi < 2; ++mi)
#pragma unroll
      for (int ni = 0; ni < 2; ++ni) {
        int j = j0 + wc + ni * 16 + jlane;
        float bsum = bi[Hc + j] + bh[Hc + j];
#pragma unroll
        for (int e = 0; e < 8; ++e)
          zz[mi][ni][e] = sigmoidf_(gi[mi][ni][e] + gh[mi][ni][e] + bsum);
      }
  }
  { // n = tanh(ig + bi + r*(hg + bh)); h' = (1-z)*n + z*h
    v8f gin[2][2] = {vzero(),vzero(),vzero(),vzero()};
    v8f ghn[2][2] = {vzero(),vzero(),vzero(),vzero()};
    mma_block(gin, sAw, Wd + (size_t)(2 * Hc + nbase) * Kin, Kin, lane);
    mma_block(ghn, sHw, Ud + (size_t)(2 * Hc + nbase) * Hc,  Hc,  lane);
#pragma unroll
    for (int mi = 0; mi < 2; ++mi)
#pragma unroll
      for (int ni = 0; ni < 2; ++ni) {
        int j = j0 + wc + ni * 16 + jlane;
        float bin = bi[2 * Hc + j], bhn = bh[2 * Hc + j];
#pragma unroll
        for (int e = 0; e < 8; ++e) {
          int b = b0 + wr + mi * 16 + ((lane >> 4) << 3) + e;
          float r = rr[mi][ni][e];
          float z = zz[mi][ni][e];
          float n = tanhf(gin[mi][ni][e] + bin + r * (ghn[mi][ni][e] + bhn));
          size_t hidx = ((size_t)dir * Bc + b) * Hc + j;
          float hnew = (1.f - z) * n + z * h32_in[hidx];
          h32_out[hidx] = hnew;
          bf16_t hb = f2bf(hnew);
          hbf_out[hidx] = hb;
          lh_out[(size_t)b * H2c + dir * Hc + j] = hb;
        }
      }
  }
}

// Layer attention: weighted = (sum_l sigmoid(gelu(lh_l@W1+b1)@w2+b2)) * out_top
__global__ void layer_attn_kernel(const bf16_t* __restrict__ lh,   // [3][B][512]
                                  const bf16_t* __restrict__ W1,   // [3][48][512]
                                  const float* __restrict__ b1,    // [3][48]
                                  const float* __restrict__ w2,    // [3][48]
                                  const float* __restrict__ b2,    // [3]
                                  bf16_t* __restrict__ ao) {       // [B][512] (this t)
  int b = blockIdx.x, tid = threadIdx.x;  // 64 threads
  __shared__ float sv[48];
  __shared__ float sscale;
  if (tid == 0) sscale = 0.f;
  __syncthreads();
  for (int l = 0; l < 3; ++l) {
    if (tid < 48) {
      const bf16_t* row = lh + ((size_t)l * Bc + b) * H2c;
      const bf16_t* w   = W1 + ((size_t)l * 48 + tid) * H2c;
      float acc = b1[l * 48 + tid];
      for (int k = 0; k < H2c; ++k) acc += bf2f(row[k]) * bf2f(w[k]);
      sv[tid] = geluf_(acc);
    }
    __syncthreads();
    if (tid == 0) {
      float s = b2[l];
      for (int j = 0; j < 48; ++j) s += sv[j] * w2[l * 48 + j];
      sscale += sigmoidf_(s);
    }
    __syncthreads();
  }
  float sc = sscale;
  const bf16_t* out = lh + ((size_t)2 * Bc + b) * H2c;
  for (int k = tid; k < H2c; k += 64) ao[(size_t)b * H2c + k] = f2bf(sc * bf2f(out[k]));
}

// logits[b][t] = silu(ao[t][b]@sa_w1.T + b1) @ sa_w2 + b2
__global__ void step_logits_kernel(const bf16_t* __restrict__ ao,   // [T][B][512]
                                   const bf16_t* __restrict__ W1,   // [64][512]
                                   const float* __restrict__ b1,    // [64]
                                   const float* __restrict__ w2,    // [64]
                                   const float* __restrict__ b2,    // [1]
                                   float* __restrict__ logits) {    // [B][T]
  int idx = blockIdx.x;
  int t = idx % Tt, b = idx / Tt;
  int j = threadIdx.x;  // 64
  const bf16_t* row = ao + ((size_t)t * Bc + b) * H2c;
  const bf16_t* w   = W1 + (size_t)j * H2c;
  float acc = b1[j];
  for (int k = 0; k < H2c; ++k) acc += bf2f(row[k]) * bf2f(w[k]);
  float v = acc * sigmoidf_(acc);  // silu
  __shared__ float red[64];
  red[j] = v * w2[j];
  __syncthreads();
  for (int s = 32; s > 0; s >>= 1) { if (j < s) red[j] += red[j + s]; __syncthreads(); }
  if (j == 0) logits[(size_t)b * Tt + t] = red[0] + b2[0];
}

// softmax over T, context, LayerNorm, FC -> out [B][2]
__global__ void finalize_kernel(const bf16_t* __restrict__ ao, const float* __restrict__ logits,
                                const float* __restrict__ ln_g, const float* __restrict__ ln_b,
                                const float* __restrict__ fc_w, const float* __restrict__ fc_b,
                                float* __restrict__ outp) {
  int b = blockIdx.x, tid = threadIdx.x;  // 256 threads
  __shared__ float swt[Tt];
  __shared__ float red[256];
  __shared__ float ctx[H2c];
  float lg = logits[(size_t)b * Tt + tid];
  red[tid] = lg; __syncthreads();
  for (int s = 128; s > 0; s >>= 1) { if (tid < s) red[tid] = fmaxf(red[tid], red[tid + s]); __syncthreads(); }
  float mx = red[0]; __syncthreads();
  float e = expf(lg - mx);
  swt[tid] = e; red[tid] = e; __syncthreads();
  for (int s = 128; s > 0; s >>= 1) { if (tid < s) red[tid] += red[tid + s]; __syncthreads(); }
  float inv = 1.f / red[0]; __syncthreads();
  float c0 = 0.f, c1 = 0.f;
  for (int t = 0; t < Tt; ++t) {
    const bf16_t* row = ao + ((size_t)t * Bc + b) * H2c;
    float wgt = swt[t] * inv;
    c0 += wgt * bf2f(row[tid]);
    c1 += wgt * bf2f(row[tid + 256]);
  }
  red[tid] = c0 + c1; __syncthreads();
  for (int s = 128; s > 0; s >>= 1) { if (tid < s) red[tid] += red[tid + s]; __syncthreads(); }
  float mu = red[0] * (1.f / H2c); __syncthreads();
  red[tid] = (c0 - mu) * (c0 - mu) + (c1 - mu) * (c1 - mu); __syncthreads();
  for (int s = 128; s > 0; s >>= 1) { if (tid < s) red[tid] += red[tid + s]; __syncthreads(); }
  float rstd = rsqrtf(red[0] * (1.f / H2c) + 1e-5f); __syncthreads();
  ctx[tid]       = (c0 - mu) * rstd * ln_g[tid]       + ln_b[tid];
  ctx[tid + 256] = (c1 - mu) * rstd * ln_g[tid + 256] + ln_b[tid + 256];
  __syncthreads();
  if (tid < 2) {
    float a = fc_b[tid];
    for (int k = 0; k < H2c; ++k) a += ctx[k] * fc_w[(size_t)tid * H2c + k];
    outp[(size_t)b * 2 + tid] = a;
  }
}

// ---------------- host ----------------
extern "C" void kernel_launch(void* const* d_in, const int* in_sizes, int n_in,
                              void* d_out, int out_size, void* d_ws, size_t ws_size,
                              hipStream_t stream) {
  (void)in_sizes; (void)n_in; (void)out_size; (void)ws_size;
  const float* x     = (const float*)d_in[0];
  const float* enc_w = (const float*)d_in[1];
  const float* enc_b = (const float*)d_in[2];
  const float* wih0  = (const float*)d_in[3];
  const float* wih   = (const float*)d_in[4];
  const float* whh   = (const float*)d_in[5];
  const float* bih   = (const float*)d_in[6];
  const float* bhh   = (const float*)d_in[7];
  const float* la_w1 = (const float*)d_in[8];
  const float* la_b1 = (const float*)d_in[9];
  const float* la_w2 = (const float*)d_in[10];
  const float* la_b2 = (const float*)d_in[11];
  const float* sa_w1 = (const float*)d_in[12];
  const float* sa_b1 = (const float*)d_in[13];
  const float* sa_w2 = (const float*)d_in[14];
  const float* sa_b2 = (const float*)d_in[15];
  const float* ln_g  = (const float*)d_in[16];
  const float* ln_b  = (const float*)d_in[17];
  const float* fc_w  = (const float*)d_in[18];
  const float* fc_b  = (const float*)d_in[19];

  char* ws = (char*)d_ws;
  size_t off = 0;
  auto alloc = [&](size_t bytes) { char* p = ws + off; off += (bytes + 255) & ~(size_t)255; return p; };

  bf16_t* seqbf  = (bf16_t*)alloc((size_t)Tt * Bc * ISc * 2);          // [T][B][128]
  bf16_t* wih0B  = (bf16_t*)alloc((size_t)2 * G3 * ISc * 2);
  bf16_t* wihB   = (bf16_t*)alloc((size_t)2 * 2 * G3 * H2c * 2);
  bf16_t* whhB   = (bf16_t*)alloc((size_t)3 * 2 * G3 * Hc * 2);
  bf16_t* laW1B  = (bf16_t*)alloc((size_t)3 * 48 * H2c * 2);
  bf16_t* saW1B  = (bf16_t*)alloc((size_t)64 * H2c * 2);
  const size_t HS = (size_t)6 * Bc * Hc;                               // per-parity h elems
  float*  h32    = (float*)alloc(2 * HS * 4);                          // [2][6][B][256]
  bf16_t* hbf    = (bf16_t*)alloc(2 * HS * 2);
  bf16_t* lhbf   = (bf16_t*)alloc((size_t)3 * Bc * H2c * 2);           // [3][B][512]
  bf16_t* ao     = (bf16_t*)alloc((size_t)Tt * Bc * H2c * 2);          // [T][B][512]
  float*  logits = (float*)alloc((size_t)Bc * Tt * 4);                 // [B][T]

  // weight conversions (cheap; re-done every call for determinism)
  auto cvt = [&](const float* s, bf16_t* d, int n) {
    to_bf16_kernel<<<(n + 255) / 256, 256, 0, stream>>>(s, d, n);
  };
  cvt(wih0,  wih0B, 2 * G3 * ISc);
  cvt(wih,   wihB,  2 * 2 * G3 * H2c);
  cvt(whh,   whhB,  3 * 2 * G3 * Hc);
  cvt(la_w1, laW1B, 3 * 48 * H2c);
  cvt(sa_w1, saW1B, 64 * H2c);

  // h0 = 0 (parity 0 is read at t=0)
  hipMemsetAsync(h32, 0, HS * 4, stream);
  hipMemsetAsync(hbf, 0, HS * 2, stream);

  // encoder -> [T][B][128] bf16
  {
    size_t n = (size_t)Bc * Tt * ISc;
    encoder_kernel<<<(unsigned)((n + 255) / 256), 256, 0, stream>>>(x, enc_w, enc_b, seqbf);
  }

  // time loop: 3 WMMA layer kernels + 1 layer-attention kernel per step
  for (int t = 0; t < Tt; ++t) {
    const int pin = t & 1, pout = pin ^ 1;
    for (int l = 0; l < 3; ++l) {
      const int Kin = (l == 0) ? ISc : H2c;
      const bf16_t* inp = (l == 0) ? seqbf + (size_t)t * Bc * ISc
                                   : lhbf + (size_t)(l - 1) * Bc * H2c;
      const bf16_t* Wih = (l == 0) ? wih0B : wihB + (size_t)(l - 1) * 2 * G3 * H2c;
      const bf16_t* Whh = whhB + (size_t)l * 2 * G3 * Hc;
      const float*  bi  = bih + (size_t)l * 2 * G3;
      const float*  bh  = bhh + (size_t)l * 2 * G3;
      const size_t lslot = (size_t)2 * l * Bc * Hc;
      const size_t smem = ((size_t)64 * Kin + (size_t)64 * Hc) * 2;
      gru_layer_kernel<<<dim3(Bc / 64, 4, 2), 128, smem, stream>>>(
          inp, Kin, Wih, Whh, bi, bh,
          hbf + pin * HS + lslot, h32 + pin * HS + lslot,
          h32 + pout * HS + lslot, hbf + pout * HS + lslot,
          lhbf + (size_t)l * Bc * H2c);
    }
    layer_attn_kernel<<<Bc, 64, 0, stream>>>(lhbf, laW1B, la_b1, la_w2, la_b2,
                                             ao + (size_t)t * Bc * H2c);
  }

  step_logits_kernel<<<Bc * Tt, 64, 0, stream>>>(ao, saW1B, sa_b1, sa_w2, sa_b2, logits);
  finalize_kernel<<<Bc, 256, 0, stream>>>(ao, logits, ln_g, ln_b, fc_w, fc_b, (float*)d_out);
}
